// MLP_GCN_32418413150598
// MI455X (gfx1250) — matrix operand order repaired and checked by reading, not compiled, not run on hardware
//
#include <hip/hip_runtime.h>
#include <hip/hip_bf16.h>
#include <cstdint>

typedef __attribute__((ext_vector_type(16))) _Float16 v16h;
typedef __attribute__((ext_vector_type(8)))  float    v8f;
typedef int ai4 __attribute__((vector_size(16)));   // matches builtin param pointee

#define TILE_M 128
#define TILE_N 128
#define TILE_K 64

#if defined(__gfx1250__) && __has_builtin(__builtin_amdgcn_global_load_async_to_lds_b128)
#define USE_ASYNC_LDS 1
#else
#define USE_ASYNC_LDS 0
#endif

#define TO_GBL(p) ((__attribute__((address_space(1))) ai4*)(uintptr_t)(p))
#define TO_LDS(p) ((__attribute__((address_space(3))) ai4*)(unsigned)(uintptr_t)(p))

// ---------------------------------------------------------------------------
// Utility kernels
// ---------------------------------------------------------------------------
__global__ void zero_f32(float* __restrict__ p, int n) {
    int i = blockIdx.x * blockDim.x + threadIdx.x;
    if (i < n) p[i] = 0.0f;
}

__global__ void degree_kernel(const int* __restrict__ src, const int* __restrict__ dst,
                              float* __restrict__ degOut, float* __restrict__ degIn, int E) {
    int e = blockIdx.x * blockDim.x + threadIdx.x;
    if (e < E) {
        atomicAdd(&degOut[src[e]], 1.0f);
        atomicAdd(&degIn[dst[e]], 1.0f);
    }
}

__global__ void rsqrt_clamp(float* __restrict__ p, int n) {
    int i = blockIdx.x * blockDim.x + threadIdx.x;
    if (i < n) p[i] = __frsqrt_rn(fmaxf(p[i], 1.0f));
}

// f32 -> f16 streaming convert, optional per-row scale (row = index/width).
__global__ void cvt_f32_f16(const float* __restrict__ in, _Float16* __restrict__ out,
                            const float* __restrict__ rowScale, int width, int nQuads) {
    int i = blockIdx.x * blockDim.x + threadIdx.x;
    if (i >= nQuads) return;
    float s = 1.0f;
    if (rowScale) s = rowScale[(i << 2) / width];
    float4 v = ((const float4*)in)[i];
    size_t o = (size_t)i << 2;
    out[o + 0] = (_Float16)(v.x * s);
    out[o + 1] = (_Float16)(v.y * s);
    out[o + 2] = (_Float16)(v.z * s);
    out[o + 3] = (_Float16)(v.w * s);
}

// W[K,N] f32 row-major  ->  Wt[N,K] f16 row-major (transpose + convert)
__global__ void cvt_transpose_w(const float* __restrict__ W, _Float16* __restrict__ Wt,
                                int K, int N) {
    int i = blockIdx.x * blockDim.x + threadIdx.x;
    if (i >= N * K) return;
    int n = i / K;
    int k = i - n * K;
    Wt[i] = (_Float16)W[(size_t)k * N + n];
}

// Gather (f16 activations, source-norm fused) + f32 scatter-add. D == 64.
// 16 threads per edge, 4 features per thread (8B vector load).
__global__ void gather_scatter64(const _Float16* __restrict__ h,
                                 const int* __restrict__ src,
                                 const int* __restrict__ dst,
                                 const float* __restrict__ normSrc,
                                 float* __restrict__ agg, int E) {
    int t = blockIdx.x * blockDim.x + threadIdx.x;
    int e = t >> 4;
    if (e >= E) return;
    int f = (t & 15) << 2;
    int s = src[e];
    int d = dst[e];
    float ns = normSrc[s];
    union { uint2 u; _Float16 x[4]; } c;
    c.u = *(const uint2*)(h + (size_t)s * 64 + f);
    float* ap = agg + (size_t)d * 64 + f;
    atomicAdd(ap + 0, (float)c.x[0] * ns);
    atomicAdd(ap + 1, (float)c.x[1] * ns);
    atomicAdd(ap + 2, (float)c.x[2] * ns);
    atomicAdd(ap + 3, (float)c.x[3] * ns);
}

// ---------------------------------------------------------------------------
// WMMA GEMM:  C[M,N] = act( A[M,K] @ Wt[N,K]^T + b[N] )
// A, Wt pre-converted f16 row-major.  Output: f16 (Ch) or f32 (Cf).
// Block tile 128x128, K-tile 64; 8 waves (wave32), each wave 32x64 = 8 WMMA accs.
// Double-buffered async-DMA staging: tile k+1 streams into LDS while tile k
// feeds the WMMA pipe.  K must be a multiple of 64 (512/256/64 here).
// ---------------------------------------------------------------------------
__global__ __launch_bounds__(256) void gemm_f16_wmma(
    const _Float16* __restrict__ A, const _Float16* __restrict__ Wt,
    const float* __restrict__ bias,
    float* __restrict__ Cf, _Float16* __restrict__ Ch,
    int M, int K, int N, int relu)
{
    __shared__ _Float16 sA[2][TILE_M][TILE_K];   // 2 x 16 KB
    __shared__ _Float16 sB[2][TILE_N][TILE_K];   // 2 x 16 KB (Wt rows = out cols)

    const int tid  = threadIdx.x;
    const int lane = tid & 31;
    const int wave = tid >> 5;
    const int wm   = (wave & 3) * 32;        // wave row offset in block tile
    const int wn   = (wave >> 2) * 64;       // wave col offset in block tile
    const int m0   = blockIdx.x * TILE_M;
    const int n0   = blockIdx.y * TILE_N;

    // Stage one 128x64-half tile pair into LDS buffer `buf` (16B chunks/lane).
    auto stage = [&](int buf, int k0) {
        #pragma unroll
        for (int it = 0; it < 4; ++it) {
            int idx = tid + it * 256;            // 0..1023
            int row = idx >> 3;
            int ho  = (idx & 7) << 3;            // half offset within row
            int gr  = m0 + row; if (gr >= M) gr = M - 1;   // clamp; OOB rows never stored
            const _Float16* gp = A + (size_t)gr * K + k0 + ho;
            _Float16* lp = &sA[buf][row][ho];
#if USE_ASYNC_LDS
            __builtin_amdgcn_global_load_async_to_lds_b128(TO_GBL(gp), TO_LDS(lp), 0, 0);
#else
            *(uint4*)lp = *(const uint4*)gp;
#endif
        }
        #pragma unroll
        for (int it = 0; it < 4; ++it) {
            int idx = tid + it * 256;
            int row = idx >> 3;
            int ho  = (idx & 7) << 3;
            int gn  = n0 + row; if (gn >= N) gn = N - 1;
            const _Float16* gp = Wt + (size_t)gn * K + k0 + ho;
            _Float16* lp = &sB[buf][row][ho];
#if USE_ASYNC_LDS
            __builtin_amdgcn_global_load_async_to_lds_b128(TO_GBL(gp), TO_LDS(lp), 0, 0);
#else
            *(uint4*)lp = *(const uint4*)gp;
#endif
        }
    };

    v8f acc[2][4] = {};
    const int nk = K / TILE_K;

    stage(0, 0);                         // prologue: prefetch first tile

    for (int kt = 0; kt < nk; ++kt) {
#if USE_ASYNC_LDS
#if __has_builtin(__builtin_amdgcn_s_wait_asynccnt)
        __builtin_amdgcn_s_wait_asynccnt(0);
#else
        asm volatile("s_wait_asynccnt 0x0" ::: "memory");
#endif
#endif
        __syncthreads();                 // tile kt resident in LDS for all waves

        if (kt + 1 < nk)                 // overlap next DMA with this compute
            stage((kt + 1) & 1, (kt + 1) * TILE_K);

        const int buf = kt & 1;

        // ---- two K=32 sub-steps per staged tile ----
        // Fragment layouts per CDNA5 ISA 7.12.2:
        //  A 16x32: lanes 0-15 -> halves 0-7=K0-7, 8-15=K16-23 (M=lane);
        //           lanes 16-31 -> halves 0-7=K8-15, 8-15=K24-31.
        //  B 32x16: N=lane%16; same K layout as A.
        const int l15 = lane & 15;
        const int kA  = (lane < 16) ? 0 : 4;     // uint offset (halves/2)
        const int kB  = (lane < 16) ? 0 : 4;

        #pragma unroll
        for (int ks = 0; ks < TILE_K; ks += 32) {
            union Frag { v16h h; unsigned u[8]; };
            Frag a[2], b[4];
            #pragma unroll
            for (int t = 0; t < 2; ++t) {
                const unsigned* p = (const unsigned*)&sA[buf][wm + l15 + 16 * t][ks];
                #pragma unroll
                for (int i = 0; i < 4; ++i) {
                    a[t].u[i]     = p[kA + i];
                    a[t].u[4 + i] = p[8 + kA + i];
                }
            }
            #pragma unroll
            for (int t = 0; t < 4; ++t) {
                const unsigned* p = (const unsigned*)&sB[buf][wn + l15 + 16 * t][ks];
                #pragma unroll
                for (int i = 0; i < 4; ++i) {
                    b[t].u[i]     = p[kB + i];
                    b[t].u[4 + i] = p[8 + kB + i];
                }
            }
            #pragma unroll
            for (int mi = 0; mi < 2; ++mi)
                #pragma unroll
                for (int ni = 0; ni < 4; ++ni)
                    acc[mi][ni] = __builtin_amdgcn_wmma_f32_16x16x32_f16(
                        false, a[mi].h, false, b[ni].h,
                        (short)0, acc[mi][ni], false, false);
        }
    }

    // ---- epilogue: bias + relu, store f16 or f32 ----
    const int l15 = lane & 15;
    #pragma unroll
    for (int mi = 0; mi < 2; ++mi) {
        int rbase = m0 + wm + mi * 16 + ((lane < 16) ? 0 : 8);
        #pragma unroll
        for (int ni = 0; ni < 4; ++ni) {
            int cn = n0 + wn + ni * 16 + l15;
            if (cn >= N) continue;
            float bb = bias[cn];
            #pragma unroll
            for (int r = 0; r < 8; ++r) {
                int row = rbase + r;
                if (row >= M) continue;
                float v = acc[mi][ni][r] + bb;
                if (relu) v = fmaxf(v, 0.0f);
                if (Ch) Ch[(size_t)row * N + cn] = (_Float16)v;
                else    Cf[(size_t)row * N + cn] = v;
            }
        }
    }
}

// ---------------------------------------------------------------------------
// Launcher
// ---------------------------------------------------------------------------
extern "C" void kernel_launch(void* const* d_in, const int* in_sizes, int n_in,
                              void* d_out, int out_size, void* d_ws, size_t ws_size,
                              hipStream_t stream) {
    const int NN = 100000;
    const int E  = 1600000;

    const float* x   = (const float*)d_in[0];
    const int*   src = (const int*)  d_in[1];
    const int*   dst = (const int*)  d_in[2];
    const float* Wf[7]  = { (const float*)d_in[3],  (const float*)d_in[5],
                            (const float*)d_in[7],  (const float*)d_in[9],
                            (const float*)d_in[11], (const float*)d_in[13],
                            (const float*)d_in[15] };
    const float* bf[7]  = { (const float*)d_in[4],  (const float*)d_in[6],
                            (const float*)d_in[8],  (const float*)d_in[10],
                            (const float*)d_in[12], (const float*)d_in[14],
                            (const float*)d_in[16] };
    const int Kdim[7] = {512, 512, 256, 64, 64, 64, 64};
    const int Ndim[7] = {512, 256,  64, 64, 64, 64, 40};

    char* ws = (char*)d_ws;
    size_t off = 0;
    auto alloc = [&](size_t bytes) -> void* {
        void* p = ws + off;
        off += (bytes + 255) & ~(size_t)255;
        return p;
    };
    float*    normSrc = (float*)alloc((size_t)NN * 4);
    float*    normDst = (float*)alloc((size_t)NN * 4);
    _Float16* X16     = (_Float16*)alloc((size_t)NN * 512 * 2);
    _Float16* H1      = (_Float16*)alloc((size_t)NN * 512 * 2);
    _Float16* H2      = (_Float16*)alloc((size_t)NN * 256 * 2);
    _Float16* H3      = (_Float16*)alloc((size_t)NN * 64 * 2);
    _Float16* H4      = (_Float16*)alloc((size_t)NN * 64 * 2);
    float*    AGG32   = (float*)alloc((size_t)NN * 64 * 4);
    _Float16* AGG16   = (_Float16*)alloc((size_t)NN * 64 * 2);
    _Float16* Wt[7];
    for (int i = 0; i < 7; ++i)
        Wt[i] = (_Float16*)alloc((size_t)Kdim[i] * Ndim[i] * 2);

    // ---- degrees -> norms (in place) ----
    zero_f32<<<(NN + 255) / 256, 256, 0, stream>>>(normSrc, NN);
    zero_f32<<<(NN + 255) / 256, 256, 0, stream>>>(normDst, NN);
    degree_kernel<<<(E + 255) / 256, 256, 0, stream>>>(src, dst, normSrc, normDst, E);
    rsqrt_clamp<<<(NN + 255) / 256, 256, 0, stream>>>(normSrc, NN);
    rsqrt_clamp<<<(NN + 255) / 256, 256, 0, stream>>>(normDst, NN);

    // ---- one-time weight transpose+convert, input convert ----
    for (int i = 0; i < 7; ++i) {
        int n = Kdim[i] * Ndim[i];
        cvt_transpose_w<<<(n + 255) / 256, 256, 0, stream>>>(Wf[i], Wt[i], Kdim[i], Ndim[i]);
    }
    {
        int q = NN * 512 / 4;
        cvt_f32_f16<<<(q + 255) / 256, 256, 0, stream>>>(x, X16, nullptr, 512, q);
    }

    auto launch_gemm = [&](const _Float16* A, int li, float* Cf, _Float16* Ch,
                           int M, int K, int N) {
        dim3 grid((M + TILE_M - 1) / TILE_M, (N + TILE_N - 1) / TILE_N);
        gemm_f16_wmma<<<grid, dim3(256), 0, stream>>>(A, Wt[li], bf[li], Cf, Ch,
                                                      M, K, N, 1);
    };

    // ---- MLP ----
    launch_gemm(X16, 0, nullptr, H1, NN, 512, 512);
    launch_gemm(H1,  1, nullptr, H2, NN, 512, 256);
    launch_gemm(H2,  2, nullptr, H3, NN, 256, 64);

    // ---- GraphConv layers ----
    auto gconv = [&](const _Float16* hin, int li, float* outF, _Float16* outH, int Nout) {
        int nAgg = NN * 64;
        zero_f32<<<(nAgg + 255) / 256, 256, 0, stream>>>(AGG32, nAgg);
        int nWork = E * 16;
        gather_scatter64<<<(nWork + 255) / 256, 256, 0, stream>>>(hin, src, dst,
                                                                  normSrc, AGG32, E);
        int q = nAgg / 4;   // fold norm_dst into the f16 conversion
        cvt_f32_f16<<<(q + 255) / 256, 256, 0, stream>>>(AGG32, AGG16, normDst, 64, q);
        launch_gemm(AGG16, li, outF, outH, NN, 64, Nout);
    };

    gconv(H3, 3, nullptr, H4, 64);
    gconv(H4, 4, nullptr, H3, 64);
    gconv(H3, 5, nullptr, H4, 64);
    gconv(H4, 6, (float*)d_out, nullptr, 40);
}
